// GPT_35081292874336
// MI455X (gfx1250) — matrix operand order repaired and checked by reading, not compiled, not run on hardware
//
#include <hip/hip_runtime.h>
#include <hip/hip_bf16.h>

// ---------------------------------------------------------------------------
// GPT forward for MI455X (gfx1250): bf16 WMMA GEMMs + flash-style WMMA attn.
// A-tiles staged with CDNA5 async global->LDS copies (ASYNCcnt), B tiles
// converted fp32->bf16 on the VALU path, next-tile global_prefetch_b8.
// ---------------------------------------------------------------------------

#define GPT_V  50257
#define GPT_S  1024
#define GPT_D  1024
#define GPT_H  16
#define GPT_L  6
#define GPT_B  2
#define GPT_HD 64
#define NTOK   (GPT_B * GPT_S)

typedef __attribute__((ext_vector_type(16))) __bf16 v16bf;
typedef __attribute__((ext_vector_type(8)))  float  v8f;

union FragBf {
    v16bf v;
    uint4 q[2];
    unsigned short s[16];
};

__device__ inline unsigned short f2bfbits(float f) {
    union { float f; unsigned int u; } c;
    c.f = f;
    unsigned int u = c.u;
    unsigned int r = u + 0x7FFFu + ((u >> 16) & 1u);  // round-to-nearest-even
    return (unsigned short)(r >> 16);
}

__device__ inline v8f wmma_bf16(const FragBf& a, const FragBf& b, v8f c) {
    return __builtin_amdgcn_wmma_f32_16x16x32_bf16(
        /*neg_a=*/false, a.v, /*neg_b=*/false, b.v,
        /*c_mod=*/(short)0, c, /*reuse_a=*/false, /*reuse_b=*/false);
}

// CDNA5 async global->LDS copy (VGLOBAL encoding, GV mode, ASYNCcnt-tracked).
// lds_off: byte offset in LDS (generic LDS pointers keep it in addr[31:0]).
__device__ inline void async_load_b128(unsigned lds_off, const void* gptr) {
    unsigned long long ga = (unsigned long long)(size_t)gptr;
    asm volatile("global_load_async_to_lds_b128 %0, %1, off"
                 :: "v"(lds_off), "v"(ga) : "memory");
}
__device__ inline void wait_asynccnt0() {
    asm volatile("s_wait_asynccnt 0x0" ::: "memory");
}

// ---------------------------------------------------------------------------
// Embedding: x[t,:] = wte[idx[t],:] + wpe[t % S,:]
// ---------------------------------------------------------------------------
__global__ __launch_bounds__(256) void embed_k(
    const int* __restrict__ idx, const float* __restrict__ wte,
    const float* __restrict__ wpe, float* __restrict__ x)
{
    const int t = blockIdx.x;
    const int tok = idx[t];
    const int pos = t & (GPT_S - 1);
    const float* we = wte + (size_t)tok * GPT_D;
    const float* pe = wpe + (size_t)pos * GPT_D;
    float* xr = x + (size_t)t * GPT_D;
    for (int i = threadIdx.x; i < GPT_D; i += 256)
        xr[i] = we[i] + pe[i];
}

// ---------------------------------------------------------------------------
// LayerNorm over D=1024, one block per token, emits bf16 (GEMM A operand).
// ---------------------------------------------------------------------------
__global__ __launch_bounds__(256) void ln_bf16(
    const float* __restrict__ x, const float* __restrict__ w,
    const float* __restrict__ bb, unsigned short* __restrict__ out)
{
    __shared__ float sb[256];
    const int t = blockIdx.x;
    const int tid = threadIdx.x;
    const float* xr = x + (size_t)t * GPT_D;

    float s = 0.f;
    for (int i = tid; i < GPT_D; i += 256) s += xr[i];
    sb[tid] = s; __syncthreads();
    for (int st = 128; st > 0; st >>= 1) {
        if (tid < st) sb[tid] += sb[tid + st];
        __syncthreads();
    }
    const float mean = sb[0] * (1.f / GPT_D);
    __syncthreads();

    float v = 0.f;
    for (int i = tid; i < GPT_D; i += 256) { float d = xr[i] - mean; v += d * d; }
    sb[tid] = v; __syncthreads();
    for (int st = 128; st > 0; st >>= 1) {
        if (tid < st) sb[tid] += sb[tid + st];
        __syncthreads();
    }
    const float rstd = rsqrtf(sb[0] * (1.f / GPT_D) + 1e-5f);

    unsigned short* orow = out + (size_t)t * GPT_D;
    for (int i = tid; i < GPT_D; i += 256)
        orow[i] = f2bfbits((xr[i] - mean) * rstd * w[i] + bb[i]);
}

// ---------------------------------------------------------------------------
// Tiled bf16 WMMA GEMM: C[M,N] = A[M,K](bf16) * B[K,N](fp32->bf16) + epilogue.
// Block tile 128x128, K-step 32, 8 waves each owning a 64x32 sub-tile.
// A tile staged via async global->LDS b128 copies; B tile converted on VALU.
// BT=1: B stored [N][K] (tied lm-head, B = wte). M%128==0, K%32==0 assumed.
// ---------------------------------------------------------------------------
__global__ __launch_bounds__(256) void gemm_wmma(
    const unsigned short* __restrict__ A, int lda,
    const float* __restrict__ Bm, int ldb, int BT,
    const float* __restrict__ bias,
    const float* __restrict__ resid,
    float* __restrict__ outF,
    unsigned short* __restrict__ outB,
    int M, int N, int K, int ldc, int gelu)
{
    __shared__ unsigned short lA[128][40];  // [m][k], +8 pad (row = 80B, 16B-aligned chunks)
    __shared__ unsigned short lB[128][40];  // [n][k] (transposed stage for contiguous K reads)

    const int tid  = threadIdx.x;
    const int lane = tid & 31;
    const int wv   = tid >> 5;
    const int wm   = wv >> 2;     // 0..1  (64 rows each)
    const int wn   = wv & 3;      // 0..3  (32 cols each)
    const int half = lane >> 4;   // lane half per ISA fragment layout
    const int l16  = lane & 15;
    const int m0 = blockIdx.y * 128;
    const int n0 = blockIdx.x * 128;

    v8f acc[4][2];
    const v8f vz = {0.f, 0.f, 0.f, 0.f, 0.f, 0.f, 0.f, 0.f};
    for (int i = 0; i < 4; ++i)
        for (int j = 0; j < 2; ++j) acc[i][j] = vz;

    const int arow = tid >> 1;    // 0..127
    const int aseg = tid & 1;     // 16 bf16 per thread per row-half

    for (int k0 = 0; k0 < K; k0 += 32) {
        // Stage A tile (bf16, 128x32) with async global->LDS copies: no VGPR
        // round-trip, frees VALU for the B-tile conversion below.
        {
            const unsigned short* gsrc = A + (size_t)(m0 + arow) * lda + k0 + aseg * 16;
            unsigned l0 = (unsigned)(size_t)&lA[arow][aseg * 16];
            async_load_b128(l0,      gsrc);
            async_load_b128(l0 + 16, gsrc + 8);
        }
        // Prefetch next B tile into GL2/WGP$ while this K-step computes.
        if (k0 + 32 < K) {
            if (!BT) {
                __builtin_prefetch(
                    &Bm[(size_t)(k0 + 32 + (tid >> 3)) * ldb + n0 + (tid & 7) * 16], 0, 1);
            } else if (n0 + (tid >> 1) < N) {
                __builtin_prefetch(
                    &Bm[(size_t)(n0 + (tid >> 1)) * ldb + k0 + 32 + (tid & 1) * 16], 0, 1);
            }
        }
        // Stage B tile (fp32 -> bf16), transposed into [n][k].
        if (!BT) {
            for (int i = 0; i < 16; ++i) {
                int e = i * 256 + tid;
                int k = e >> 7, n = e & 127;
                float v = (n0 + n < N) ? Bm[(size_t)(k0 + k) * ldb + (n0 + n)] : 0.f;
                lB[n][k] = f2bfbits(v);
            }
        } else {
            for (int i = 0; i < 16; ++i) {
                int e = i * 256 + tid;
                int n = e >> 5, k = e & 31;
                float v = (n0 + n < N) ? Bm[(size_t)(n0 + n) * ldb + (k0 + k)] : 0.f;
                lB[n][k] = f2bfbits(v);
            }
        }
        wait_asynccnt0();     // A-tile async copies complete (ASYNCcnt == 0)
        __syncthreads();

        // Build fragments (ISA 7.12.2 16-bit layouts) and issue 8 WMMAs.
        FragBf af[4], bfr[2];
        for (int mi = 0; mi < 4; ++mi) {
            int row = wm * 64 + mi * 16 + l16;
            af[mi].q[0] = *(const uint4*)&lA[row][half * 8];        // K {0..7}/{8..15}
            af[mi].q[1] = *(const uint4*)&lA[row][16 + half * 8];   // K {16..23}/{24..31}
        }
        for (int ni = 0; ni < 2; ++ni) {
            int col = wn * 32 + ni * 16 + l16;
            bfr[ni].q[0] = *(const uint4*)&lB[col][half * 8];       // K {0..7}/{8..15}
            bfr[ni].q[1] = *(const uint4*)&lB[col][16 + half * 8];  // K {16..23}/{24..31}
        }
        for (int mi = 0; mi < 4; ++mi)
            for (int ni = 0; ni < 2; ++ni)
                acc[mi][ni] = wmma_bf16(af[mi], bfr[ni], acc[mi][ni]);
        __syncthreads();
    }

    // Epilogue: bias / GELU(exact) / residual / fp32 + bf16 stores.
    for (int mi = 0; mi < 4; ++mi) {
        for (int ni = 0; ni < 2; ++ni) {
            int colg = n0 + wn * 32 + ni * 16 + l16;
            if (colg >= N) continue;
            for (int r = 0; r < 8; ++r) {
                int rowg = m0 + wm * 64 + mi * 16 + half * 8 + r;
                float v = acc[mi][ni][r];
                if (bias)  v += bias[colg];
                if (gelu)  v = 0.5f * v * (1.f + erff(v * 0.70710678118654752f));
                if (resid) v += resid[(size_t)rowg * ldc + colg];
                if (outF)  outF[(size_t)rowg * ldc + colg] = v;
                if (outB)  outB[(size_t)rowg * ldc + colg] = f2bfbits(v);
            }
        }
    }
}

// ---------------------------------------------------------------------------
// Causal attention, one wave per (b, h, 16-query tile). QK^T and P*V on WMMA,
// score rows (16 x S fp32) resident in LDS, wave-cooperative softmax.
// qkv layout: [tok][3D] bf16 with q|k|v at offsets 0|D|2D, head h at h*64.
// ---------------------------------------------------------------------------
__global__ __launch_bounds__(32) void attn_wmma(
    const unsigned short* __restrict__ qkv, unsigned short* __restrict__ y)
{
    __shared__ float sc[16][GPT_S];   // 64 KB of the 320 KB WGP LDS

    const int qt   = blockIdx.x;      // query tile 0..63
    const int h    = blockIdx.y;
    const int b    = blockIdx.z;
    const int lane = threadIdx.x;
    const int half = lane >> 4;
    const int l16  = lane & 15;

    // Q fragments (A operand, 16 queries x 64 dims -> 2 K-chunks of 32).
    FragBf qf[2];
    {
        const unsigned short* qp =
            qkv + (size_t)(b * GPT_S + qt * 16 + l16) * (3 * GPT_D) + h * GPT_HD;
        for (int kk = 0; kk < 2; ++kk) {
            qf[kk].q[0] = *(const uint4*)(qp + kk * 32 + half * 8);
            qf[kk].q[1] = *(const uint4*)(qp + kk * 32 + 16 + half * 8);
        }
    }

    const v8f vz = {0.f, 0.f, 0.f, 0.f, 0.f, 0.f, 0.f, 0.f};

    // Phase 1: scores for all causal key tiles.
    for (int kt = 0; kt <= qt; ++kt) {
        const unsigned short* kp =
            qkv + (size_t)(b * GPT_S + kt * 16 + l16) * (3 * GPT_D) + GPT_D + h * GPT_HD;
        FragBf kf0, kf1;  // B operand = K^T: lane holds key column, contiguous dims
        kf0.q[0] = *(const uint4*)(kp + half * 8);
        kf0.q[1] = *(const uint4*)(kp + 16 + half * 8);
        kf1.q[0] = *(const uint4*)(kp + 32 + half * 8);
        kf1.q[1] = *(const uint4*)(kp + 32 + 16 + half * 8);

        v8f s = vz;
        s = wmma_bf16(qf[0], kf0, s);
        s = wmma_bf16(qf[1], kf1, s);

        for (int r = 0; r < 8; ++r) {
            int m  = half * 8 + r;          // query row in tile
            int kg = kt * 16 + l16;         // global key
            float v = s[r] * 0.125f;        // 1/sqrt(64)
            if (kg > qt * 16 + m) v = -3.0e38f;  // causal mask (diagonal tile)
            sc[m][kg] = v;
        }
    }
    __syncthreads();

    // Phase 2: softmax per query row over valid keys.
    const int ncols = (qt + 1) * 16;
    for (int m = 0; m < 16; ++m) {
        const int cnt = qt * 16 + m + 1;
        float mx = -3.0e38f;
        for (int c = lane; c < cnt; c += 32) mx = fmaxf(mx, sc[m][c]);
        for (int o = 16; o > 0; o >>= 1) mx = fmaxf(mx, __shfl_xor(mx, o, 32));
        float sum = 0.f;
        for (int c = lane; c < cnt; c += 32) {
            float e = __expf(sc[m][c] - mx);
            sc[m][c] = e;
            sum += e;
        }
        for (int o = 16; o > 0; o >>= 1) sum += __shfl_xor(sum, o, 32);
        const float inv = 1.f / sum;
        for (int c = lane; c < ncols; c += 32)
            sc[m][c] = (c < cnt) ? sc[m][c] * inv : 0.f;
    }
    __syncthreads();

    // Phase 3: y = P @ V via WMMA (A = P rows re-quantized to bf16).
    v8f yacc[4];
    for (int d = 0; d < 4; ++d) yacc[d] = vz;
    const int nch = (ncols + 31) >> 5;
    for (int ch = 0; ch < nch; ++ch) {
        FragBf pf;
        for (int e = 0; e < 16; ++e) {
            int kk = (e < 8) ? (half * 8 + e) : (16 + half * 8 + (e - 8));
            int kg = ch * 32 + kk;
            float v = (kg < ncols) ? sc[l16][kg] : 0.f;
            pf.s[e] = f2bfbits(v);
        }
        for (int d = 0; d < 4; ++d) {
            FragBf vf;  // B operand: lane holds dim column d*16+l16, keys along K
            for (int e = 0; e < 16; ++e) {
                int key = ch * 32 + ((e < 8) ? (half * 8 + e) : (16 + half * 8 + (e - 8)));  // < S always (rounded tile)
                vf.s[e] = qkv[(size_t)(b * GPT_S + key) * (3 * GPT_D)
                              + 2 * GPT_D + h * GPT_HD + d * 16 + l16];
            }
            yacc[d] = wmma_bf16(pf, vf, yacc[d]);
        }
    }

    // Store y tile (bf16, feeds the proj GEMM).
    for (int d = 0; d < 4; ++d)
        for (int r = 0; r < 8; ++r) {
            int m = half * 8 + r;
            y[(size_t)(b * GPT_S + qt * 16 + m) * GPT_D + h * GPT_HD + d * 16 + l16] =
                f2bfbits(yacc[d][r]);
        }
}

// ---------------------------------------------------------------------------
// Host-side orchestration (graph-capture safe: only launches on `stream`).
// ---------------------------------------------------------------------------
extern "C" void kernel_launch(void* const* d_in, const int* in_sizes, int n_in,
                              void* d_out, int out_size, void* d_ws, size_t ws_size,
                              hipStream_t stream)
{
    (void)in_sizes; (void)n_in; (void)out_size; (void)ws_size;

    const int*   idx    = (const int*)  d_in[0];
    const float* wte    = (const float*)d_in[1];
    const float* wpe    = (const float*)d_in[2];
    const float* ln1_w  = (const float*)d_in[3];
    const float* ln1_b  = (const float*)d_in[4];
    const float* attn_w = (const float*)d_in[5];
    const float* attn_b = (const float*)d_in[6];
    const float* proj_w = (const float*)d_in[7];
    const float* proj_b = (const float*)d_in[8];
    const float* ln2_w  = (const float*)d_in[9];
    const float* ln2_b  = (const float*)d_in[10];
    const float* fc1_w  = (const float*)d_in[11];
    const float* fc1_b  = (const float*)d_in[12];
    const float* fc2_w  = (const float*)d_in[13];
    const float* fc2_b  = (const float*)d_in[14];
    const float* lnf_w  = (const float*)d_in[15];
    const float* lnf_b  = (const float*)d_in[16];

    // Workspace layout (44 MiB total):
    char* ws = (char*)d_ws;
    float*          x    = (float*)ws;                                       // 8 MiB residual
    unsigned short* hbuf = (unsigned short*)(ws + (size_t) 8 * 1024 * 1024); // 4 MiB LN out
    unsigned short* qkv  = (unsigned short*)(ws + (size_t)12 * 1024 * 1024); // 12 MiB
    unsigned short* ybuf = (unsigned short*)(ws + (size_t)24 * 1024 * 1024); // 4 MiB
    unsigned short* act  = (unsigned short*)(ws + (size_t)28 * 1024 * 1024); // 16 MiB

    embed_k<<<NTOK, 256, 0, stream>>>(idx, wte, wpe, x);

    for (int l = 0; l < GPT_L; ++l) {
        const size_t Dl = (size_t)l * GPT_D;

        // LN1 -> h
        ln_bf16<<<NTOK, 256, 0, stream>>>(x, ln1_w + Dl, ln1_b + Dl, hbuf);
        // qkv = h @ attn_w + attn_b  (bf16 out)
        gemm_wmma<<<dim3(3 * GPT_D / 128, NTOK / 128), 256, 0, stream>>>(
            hbuf, GPT_D, attn_w + Dl * 3 * GPT_D, 3 * GPT_D, 0,
            attn_b + (size_t)l * 3 * GPT_D, nullptr, nullptr, qkv,
            NTOK, 3 * GPT_D, GPT_D, 3 * GPT_D, 0);
        // attention -> ybuf
        attn_wmma<<<dim3(GPT_S / 16, GPT_H, GPT_B), 32, 0, stream>>>(qkv, ybuf);
        // x += ybuf @ proj_w + proj_b
        gemm_wmma<<<dim3(GPT_D / 128, NTOK / 128), 256, 0, stream>>>(
            ybuf, GPT_D, proj_w + Dl * GPT_D, GPT_D, 0,
            proj_b + Dl, x, x, nullptr,
            NTOK, GPT_D, GPT_D, GPT_D, 0);
        // LN2 -> h
        ln_bf16<<<NTOK, 256, 0, stream>>>(x, ln2_w + Dl, ln2_b + Dl, hbuf);
        // act = gelu(h @ fc1_w + fc1_b)  (bf16 out)
        gemm_wmma<<<dim3(4 * GPT_D / 128, NTOK / 128), 256, 0, stream>>>(
            hbuf, GPT_D, fc1_w + Dl * 4 * GPT_D, 4 * GPT_D, 0,
            fc1_b + (size_t)l * 4 * GPT_D, nullptr, nullptr, act,
            NTOK, 4 * GPT_D, GPT_D, 4 * GPT_D, 1);
        // x += act @ fc2_w + fc2_b
        gemm_wmma<<<dim3(GPT_D / 128, NTOK / 128), 256, 0, stream>>>(
            act, 4 * GPT_D, fc2_w + (size_t)l * 4 * GPT_D * GPT_D, GPT_D, 0,
            fc2_b + Dl, x, x, nullptr,
            NTOK, GPT_D, 4 * GPT_D, GPT_D, 0);
    }

    // Final LN -> h, then logits = h @ wte^T (transposed-B path, N-guarded).
    ln_bf16<<<NTOK, 256, 0, stream>>>(x, lnf_w, lnf_b, hbuf);
    gemm_wmma<<<dim3((GPT_V + 127) / 128, NTOK / 128), 256, 0, stream>>>(
        hbuf, GPT_D, wte, GPT_D, 1,
        nullptr, nullptr, (float*)d_out, nullptr,
        NTOK, GPT_V, GPT_D, GPT_V, 0);
}
